// GRFExactAttention_50173807952660
// MI455X (gfx1250) — matrix operand order repaired and checked by reading, not compiled, not run on hardware
//
#include <hip/hip_runtime.h>

#define BB 8
#define NN 1024
#define CC 768
#define HH 12
#define DD 64
#define EPSF 1e-6f

typedef __attribute__((ext_vector_type(16))) __bf16 v16bf;
typedef __attribute__((ext_vector_type(8)))  __bf16 v8bf;
typedef __attribute__((ext_vector_type(8)))  float  v8f;

// ---- helpers ----
static __device__ __forceinline__ v16bf cat8(v8bf lo, v8bf hi) {
  return __builtin_shufflevector(lo, hi, 0, 1, 2, 3, 4, 5, 6, 7, 8, 9, 10, 11, 12,
                                 13, 14, 15);
}
static __device__ __forceinline__ v16bf cvt16(float4 f0, float4 f1, float4 f2,
                                              float4 f3) {
  v16bf a;
  a[0] = (__bf16)f0.x;  a[1] = (__bf16)f0.y;  a[2] = (__bf16)f0.z;  a[3] = (__bf16)f0.w;
  a[4] = (__bf16)f1.x;  a[5] = (__bf16)f1.y;  a[6] = (__bf16)f1.z;  a[7] = (__bf16)f1.w;
  a[8] = (__bf16)f2.x;  a[9] = (__bf16)f2.y;  a[10] = (__bf16)f2.z; a[11] = (__bf16)f2.w;
  a[12] = (__bf16)f3.x; a[13] = (__bf16)f3.y; a[14] = (__bf16)f3.z; a[15] = (__bf16)f3.w;
  return a;
}

// ---- tile loaders: gfx1250 WMMA VGPR layouts, contiguous 16B per-lane loads ----
// A 16x32 (MxK) from bf16 row-major: lane<16 row=lane, K 0-7 + 16-23;
// lane>=16 row=lane-16, K 8-15 + 24-31.
static __device__ __forceinline__ v16bf load_A_bf(const __bf16* base, int ld,
                                                  int lane) {
  int r = lane & 15, kh = (lane >> 4) << 3;
  const __bf16* p = base + (size_t)r * ld + kh;
  v8bf lo = *(const v8bf*)p;
  v8bf hi = *(const v8bf*)(p + 16);
  return cat8(lo, hi);
}
// B 32x16 (KxN) where the source stores B^T row-major (per output column, a
// contiguous run over k): lane<16 col=lane K 0-7 + 16-23; lane>=16 col=lane-16 K 8-15 + 24-31.
static __device__ __forceinline__ v16bf load_B_ct(const __bf16* base, int ld,
                                                  int lane) {
  int c = lane & 15, kb = (lane >> 4) << 3;
  const __bf16* p = base + (size_t)c * ld + kb;
  v8bf lo = *(const v8bf*)p;
  v8bf hi = *(const v8bf*)(p + 16);
  return cat8(lo, hi);
}
static __device__ __forceinline__ v8f wmma_bf16(v16bf a, v16bf b, v8f c) {
  return __builtin_amdgcn_wmma_f32_16x16x32_bf16(false, a, false, b, (short)0, c,
                                                 false, false);
}

// ---- stage 0a: elementwise fp32 -> bf16 (for x) ----
__global__ __launch_bounds__(256) void k_cvt_bf(const float* __restrict__ src,
                                                __bf16* __restrict__ dst,
                                                size_t n) {
  size_t i = ((size_t)blockIdx.x * blockDim.x + threadIdx.x) * 8;
  if (i >= n) return;
  float4 f0 = *(const float4*)(src + i);
  float4 f1 = *(const float4*)(src + i + 4);
  v8bf o;
  o[0] = (__bf16)f0.x; o[1] = (__bf16)f0.y; o[2] = (__bf16)f0.z; o[3] = (__bf16)f0.w;
  o[4] = (__bf16)f1.x; o[5] = (__bf16)f1.y; o[6] = (__bf16)f1.z; o[7] = (__bf16)f1.w;
  *(v8bf*)(dst + i) = o;
}

// ---- stage 0b: fp32 -> bf16 transpose (LDS-tiled, coalesced both sides) ----
__global__ __launch_bounds__(256) void k_transpose_bf(const float* __restrict__ src,
                                                      __bf16* __restrict__ dst,
                                                      int R, int C) {
  __shared__ __bf16 tile[32][33];
  int bx = blockIdx.x * 32, by = blockIdx.y * 32;
  int tx = threadIdx.x, ty = threadIdx.y;  // blockDim = (32, 8)
#pragma unroll
  for (int j = 0; j < 32; j += 8) {
    int r = by + ty + j, c = bx + tx;
    tile[ty + j][tx] = (__bf16)src[(size_t)r * C + c];
  }
  __syncthreads();
#pragma unroll
  for (int j = 0; j < 32; j += 8) {
    int c = bx + ty + j, r = by + tx;
    dst[(size_t)c * R + r] = tile[tx][ty + j];
  }
}

// ---- stage 1: qkv = x @ w_qkv, relu+eps on q/k; write bf16 transposed [B,H,D,N]
// one wave computes a 32x64 block: 2 row-tiles x 4 col-tiles; every B tile
// feeds two WMMAs, every A tile feeds four.
__global__ __launch_bounds__(256) void k_qkv(const __bf16* __restrict__ xbf,
                                             const __bf16* __restrict__ wqkvT,
                                             __bf16* __restrict__ qT,
                                             __bf16* __restrict__ kT,
                                             __bf16* __restrict__ vT) {
  const int NG = (3 * CC) / 64;  // 36 column groups of 64
  int tid = blockIdx.x * blockDim.x + threadIdx.x;
  int w = tid >> 5, lane = tid & 31;
  int mt = w / NG, cg = w % NG;
  if (mt >= (BB * NN) / 32) return;
  int row0 = mt * 32, col0 = cg * 64;
  v8f acc0[4] = {{}, {}, {}, {}};
  v8f acc1[4] = {{}, {}, {}, {}};
  for (int k0 = 0; k0 < CC; k0 += 32) {
    v16bf a0 = load_A_bf(xbf + (size_t)row0 * CC + k0, CC, lane);
    v16bf a1 = load_A_bf(xbf + (size_t)(row0 + 16) * CC + k0, CC, lane);
#pragma unroll
    for (int j = 0; j < 4; ++j) {
      v16bf b = load_B_ct(wqkvT + (size_t)(col0 + 16 * j) * CC + k0, CC, lane);
      acc0[j] = wmma_bf16(a0, b, acc0[j]);
      acc1[j] = wmma_bf16(a1, b, acc1[j]);
    }
  }
  // col0 is 64-aligned: part and head are uniform across the strip, d = 16j + c
  int part = col0 / CC, cc0 = col0 % CC, h = cc0 / DD;
  __bf16* dst = (part == 0) ? qT : (part == 1) ? kT : vT;
  int c = lane & 15, rbase = (lane >> 4) << 3;
#pragma unroll
  for (int rh = 0; rh < 2; ++rh) {
    int row = row0 + 16 * rh + rbase;
    int bIdx = row / NN, nbase = row % NN;  // 8 consecutive n, same batch
    const v8f* accs = rh ? acc1 : acc0;
#pragma unroll
    for (int j = 0; j < 4; ++j) {
      v8bf vs;
#pragma unroll
      for (int i = 0; i < 8; ++i) {
        float val = accs[j][i];
        if (part < 2) val = fmaxf(val, 0.0f) + EPSF;
        vs[i] = (__bf16)val;
      }
      *(v8bf*)(dst + (((size_t)bIdx * HH + h) * DD + 16 * j + c) * NN + nbase) = vs;
    }
  }
}

// ---- stage 2: Qs[m,d] = Q[m,d] + 0.1 * sum_n mask[n,m] Q[n,d]  (per b,h) ----
// one wave computes a 32 x 64 block (all of D); every Q B-tile feeds two WMMAs
__global__ __launch_bounds__(256) void k_smooth(const __bf16* __restrict__ maskT,
                                                const __bf16* __restrict__ QT,
                                                __bf16* __restrict__ Qs) {
  int tid = blockIdx.x * blockDim.x + threadIdx.x;
  int w = tid >> 5, lane = tid & 31;
  int bh = w / (NN / 32);
  if (bh >= BB * HH) return;
  int mt = w % (NN / 32);
  const __bf16* Qt = QT + (size_t)bh * DD * NN;
  __bf16* O = Qs + (size_t)bh * NN * DD;
  int m0 = mt * 32;
  v8f acc0[4] = {{}, {}, {}, {}};
  v8f acc1[4] = {{}, {}, {}, {}};
  for (int n0 = 0; n0 < NN; n0 += 32) {
    v16bf a0 = load_A_bf(maskT + (size_t)m0 * NN + n0, NN, lane);
    v16bf a1 = load_A_bf(maskT + (size_t)(m0 + 16) * NN + n0, NN, lane);
#pragma unroll
    for (int j = 0; j < 4; ++j) {
      v16bf b = load_B_ct(Qt + (size_t)(16 * j) * NN + n0, NN, lane);
      acc0[j] = wmma_bf16(a0, b, acc0[j]);
      acc1[j] = wmma_bf16(a1, b, acc1[j]);
    }
  }
  int c = lane & 15, rbase = (lane >> 4) << 3;
#pragma unroll
  for (int rh = 0; rh < 2; ++rh) {
    const v8f* accs = rh ? acc1 : acc0;
#pragma unroll
    for (int j = 0; j < 4; ++j) {
#pragma unroll
      for (int i = 0; i < 8; ++i) {
        int m = m0 + 16 * rh + rbase + i;
        float base = (float)Qt[(size_t)(16 * j + c) * NN + m];
        O[(size_t)m * DD + 16 * j + c] = (__bf16)(base + 0.1f * accs[j][i]);
      }
    }
  }
}

// ---- stage 3: fused masked linear attention per (b,h), 16 q-rows per wave ----
__global__ __launch_bounds__(256) void k_attn(const float* __restrict__ mask,
                                              const __bf16* __restrict__ Qs,
                                              const __bf16* __restrict__ Ks,
                                              const __bf16* __restrict__ vT,
                                              __bf16* __restrict__ attn) {
  __shared__ float sbuf[8][16][36];  // per-wave 16x32 S patch, 16B-aligned rows
  int tid = blockIdx.x * blockDim.x + threadIdx.x;
  int w = tid >> 5, lane = tid & 31;
  int wib = threadIdx.x >> 5;
  int bh = w / (NN / 16), nt = w % (NN / 16);
  if (bh >= BB * HH) return;
  const __bf16* Qp = Qs + (size_t)bh * NN * DD;
  const __bf16* Kp = Ks + (size_t)bh * NN * DD;
  const __bf16* Vt = vT + (size_t)bh * DD * NN;
  int n0 = nt * 16;
  v16bf aq0 = load_A_bf(Qp + (size_t)n0 * DD + 0, DD, lane);
  v16bf aq1 = load_A_bf(Qp + (size_t)n0 * DD + 32, DD, lane);
  v8f o0 = {}, o1 = {}, o2 = {}, o3 = {};
  float rsum[8];
#pragma unroll
  for (int i = 0; i < 8; ++i) rsum[i] = 0.0f;
  int c = lane & 15, rbase = (lane >> 4) << 3;

  for (int m0 = 0; m0 < NN; m0 += 32) {
    // S[n,m] = sum_d q'[n,d] k'[m,d]; B^T runs are rows of Ks (contiguous)
    v8f s0 = {}, s1 = {};
    s0 = wmma_bf16(aq0, load_B_ct(Kp + (size_t)m0 * DD + 0, DD, lane), s0);
    s0 = wmma_bf16(aq1, load_B_ct(Kp + (size_t)m0 * DD + 32, DD, lane), s0);
    s1 = wmma_bf16(aq0, load_B_ct(Kp + (size_t)(m0 + 16) * DD + 0, DD, lane), s1);
    s1 = wmma_bf16(aq1, load_B_ct(Kp + (size_t)(m0 + 16) * DD + 32, DD, lane), s1);
    // mask multiply in C-layout regs, row-sum accumulation, stage S to LDS
#pragma unroll
    for (int i = 0; i < 8; ++i) {
      int n = n0 + rbase + i;
      float t0 = s0[i] * mask[(size_t)n * NN + m0 + c];
      float t1 = s1[i] * mask[(size_t)n * NN + m0 + 16 + c];
      rsum[i] += t0 + t1;
      sbuf[wib][rbase + i][c] = t0;
      sbuf[wib][rbase + i][16 + c] = t1;
    }
    // reload masked S as a 16x32 bf16 A-tile (same-wave DS ops are in-order)
    v16bf as;
    {
      int r = lane & 15, kh = (lane >> 4) << 3;
      const float* rp = &sbuf[wib][r][0];
      float4 f0 = *(const float4*)(rp + kh);
      float4 f1 = *(const float4*)(rp + kh + 4);
      float4 f2 = *(const float4*)(rp + kh + 16);
      float4 f3 = *(const float4*)(rp + kh + 20);
      as = cvt16(f0, f1, f2, f3);
    }
    // O[16x64] += S_masked @ V[m0:m0+32, :]; B^T runs are rows of vT
    o0 = wmma_bf16(as, load_B_ct(Vt + (size_t)0 * NN + m0, NN, lane), o0);
    o1 = wmma_bf16(as, load_B_ct(Vt + (size_t)16 * NN + m0, NN, lane), o1);
    o2 = wmma_bf16(as, load_B_ct(Vt + (size_t)32 * NN + m0, NN, lane), o2);
    o3 = wmma_bf16(as, load_B_ct(Vt + (size_t)48 * NN + m0, NN, lane), o3);
  }
  // reduce row sums across the 16 lanes of each half (rows match C/D striping)
#pragma unroll
  for (int i = 0; i < 8; ++i) {
    float r = rsum[i];
    r += __shfl_xor(r, 1, 32);
    r += __shfl_xor(r, 2, 32);
    r += __shfl_xor(r, 4, 32);
    r += __shfl_xor(r, 8, 32);
    rsum[i] = 1.0f / (r + EPSF);
  }
  int b = bh / HH, h = bh % HH;
#pragma unroll
  for (int i = 0; i < 8; ++i) {
    int n = n0 + rbase + i;
    size_t rowbase = ((size_t)b * NN + n) * CC + (size_t)h * DD;
    attn[rowbase + 0 + c]  = (__bf16)(o0[i] * rsum[i]);
    attn[rowbase + 16 + c] = (__bf16)(o1[i] * rsum[i]);
    attn[rowbase + 32 + c] = (__bf16)(o2[i] * rsum[i]);
    attn[rowbase + 48 + c] = (__bf16)(o3[i] * rsum[i]);
  }
}

// ---- stage 4: out = attn @ w_out + b_out; 32x64 block per wave ----
__global__ __launch_bounds__(256) void k_proj(const __bf16* __restrict__ attn,
                                              const __bf16* __restrict__ woutT,
                                              const float* __restrict__ b_out,
                                              float* __restrict__ out) {
  const int NG = CC / 64;  // 12 column groups
  int tid = blockIdx.x * blockDim.x + threadIdx.x;
  int w = tid >> 5, lane = tid & 31;
  int mt = w / NG, cg = w % NG;
  if (mt >= (BB * NN) / 32) return;
  int row0 = mt * 32, col0 = cg * 64;
  v8f acc0[4] = {{}, {}, {}, {}};
  v8f acc1[4] = {{}, {}, {}, {}};
  for (int k0 = 0; k0 < CC; k0 += 32) {
    v16bf a0 = load_A_bf(attn + (size_t)row0 * CC + k0, CC, lane);
    v16bf a1 = load_A_bf(attn + (size_t)(row0 + 16) * CC + k0, CC, lane);
#pragma unroll
    for (int j = 0; j < 4; ++j) {
      v16bf b = load_B_ct(woutT + (size_t)(col0 + 16 * j) * CC + k0, CC, lane);
      acc0[j] = wmma_bf16(a0, b, acc0[j]);
      acc1[j] = wmma_bf16(a1, b, acc1[j]);
    }
  }
  int c = lane & 15, rbase = (lane >> 4) << 3;
#pragma unroll
  for (int rh = 0; rh < 2; ++rh) {
    const v8f* accs = rh ? acc1 : acc0;
#pragma unroll
    for (int j = 0; j < 4; ++j) {
      float bias = b_out[col0 + 16 * j + c];
#pragma unroll
      for (int i = 0; i < 8; ++i)
        out[(size_t)(row0 + 16 * rh + rbase + i) * CC + col0 + 16 * j + c] =
            accs[j][i] + bias;
    }
  }
}

extern "C" void kernel_launch(void* const* d_in, const int* in_sizes, int n_in,
                              void* d_out, int out_size, void* d_ws, size_t ws_size,
                              hipStream_t stream) {
  const float* x     = (const float*)d_in[0];
  const float* mask  = (const float*)d_in[1];
  const float* w_qkv = (const float*)d_in[2];
  const float* w_out = (const float*)d_in[3];
  const float* b_out = (const float*)d_in[4];
  float* out = (float*)d_out;

  const size_t S = (size_t)BB * HH * NN * DD;  // per-tensor element count
  char* ws = (char*)d_ws;
  size_t off = 0;
  __bf16* qT    = (__bf16*)(ws + off); off += S * 2;
  __bf16* kT    = (__bf16*)(ws + off); off += S * 2;
  __bf16* vT    = (__bf16*)(ws + off); off += S * 2;
  __bf16* qs    = (__bf16*)(ws + off); off += S * 2;
  __bf16* ks    = (__bf16*)(ws + off); off += S * 2;
  __bf16* attn  = (__bf16*)(ws + off); off += S * 2;
  __bf16* xbf   = (__bf16*)(ws + off); off += S * 2;
  __bf16* maskT = (__bf16*)(ws + off); off += (size_t)NN * NN * 2;
  __bf16* wqkvT = (__bf16*)(ws + off); off += (size_t)CC * 3 * CC * 2;
  __bf16* woutT = (__bf16*)(ws + off); off += (size_t)CC * CC * 2;

  // stage 0: bf16 conversions (x elementwise; weights/mask transposed)
  {
    size_t n = S;  // B*N*C
    int blocks = (int)((n / 8 + 255) / 256);
    k_cvt_bf<<<blocks, 256, 0, stream>>>(x, xbf, n);
    dim3 blk(32, 8);
    k_transpose_bf<<<dim3(NN / 32, NN / 32), blk, 0, stream>>>(mask, maskT, NN, NN);
    k_transpose_bf<<<dim3((3 * CC) / 32, CC / 32), blk, 0, stream>>>(w_qkv, wqkvT,
                                                                     CC, 3 * CC);
    k_transpose_bf<<<dim3(CC / 32, CC / 32), blk, 0, stream>>>(w_out, woutT, CC, CC);
  }
  {
    int waves = ((BB * NN) / 32) * ((3 * CC) / 64);
    int blocks = (waves * 32 + 255) / 256;
    k_qkv<<<blocks, 256, 0, stream>>>(xbf, wqkvT, qT, kT, vT);
  }
  {
    int waves = BB * HH * (NN / 32);
    int blocks = (waves * 32 + 255) / 256;
    k_smooth<<<blocks, 256, 0, stream>>>(maskT, qT, qs);
    k_smooth<<<blocks, 256, 0, stream>>>(maskT, kT, ks);
  }
  {
    int waves = BB * HH * (NN / 16);
    int blocks = (waves * 32 + 255) / 256;
    k_attn<<<blocks, 256, 0, stream>>>(mask, qs, ks, vT, attn);
  }
  {
    int waves = ((BB * NN) / 32) * (CC / 64);
    int blocks = (waves * 32 + 255) / 256;
    k_proj<<<blocks, 256, 0, stream>>>(attn, woutT, b_out, out);
  }
}
